// GT_50955492000030
// MI455X (gfx1250) — compile-verified, matrix operand order checked by reading, not run on hardware
//
#include <hip/hip_runtime.h>
#include <hip/hip_bf16.h>
#include <stddef.h>

// ---------------- problem constants ----------------
#define DIMN   768
#define HEADS  12
#define HDIM   64
#define BB     2
#define SS     2048
#define NTOK   (BB*SS)          // 4096 token rows
#define EPSF   1e-5f

typedef __attribute__((ext_vector_type(16))) _Float16 v16h;
typedef __attribute__((ext_vector_type(8)))  _Float16 v8h;
typedef __attribute__((ext_vector_type(8)))  float    v8f;

// ---------------- WMMA helpers ----------------
__device__ __forceinline__ v8f wmma_f16(v16h a, v16h b, v8f c) {
  // D = A(16x32 f16) * B(32x16 f16) + C(16x16 f32)
  return __builtin_amdgcn_wmma_f32_16x16x32_f16(
      false, a, false, b, (short)0, c, false, false);
}

// Load one 16-lane fragment row per the CDNA5 16-bit A/B VGPR layout:
// lane L (<16) holds row L, K = kc+[0..7] and kc+[16..23];
// lane L+16 holds row L, K = kc+[8..15] and kc+[24..31].
// Both halves are contiguous 16B loads.
__device__ __forceinline__ v16h ld_frag(const _Float16* __restrict__ base,
                                        int stride, int row, int kc, int lane) {
  const _Float16* p = base + (size_t)row * stride + kc + ((lane >> 4) << 3);
  v8h lo = *(const v8h*)(p);
  v8h hi = *(const v8h*)(p + 16);
  v16h r;
#pragma unroll
  for (int i = 0; i < 8; ++i) { r[i] = lo[i]; r[i + 8] = hi[i]; }
  return r;
}

// ---------------- weight transpose + cast (w[k][n] -> wt[n][k] f16) ----------------
__global__ void k_transpose_cast(const float* __restrict__ w, _Float16* __restrict__ wt) {
  __shared__ float tile[32][33];
  int bx = blockIdx.x % (DIMN / 32);
  int by = blockIdx.x / (DIMN / 32);
  int tx = threadIdx.x & 31, ty = threadIdx.x >> 5;   // 32 x 8
#pragma unroll
  for (int i = ty; i < 32; i += 8)
    tile[i][tx] = w[(size_t)(by * 32 + i) * DIMN + bx * 32 + tx];
  __syncthreads();
#pragma unroll
  for (int i = ty; i < 32; i += 8)
    wt[(size_t)(bx * 32 + i) * DIMN + by * 32 + tx] = (_Float16)tile[tx][i];
}

// ---------------- layernorm (fp32 in -> f16 out), one block per token row ----------------
__global__ void k_layernorm(const float* __restrict__ x, const float* __restrict__ w,
                            const float* __restrict__ b, _Float16* __restrict__ out) {
  __shared__ float sred[256];
  __shared__ float s_mu, s_rstd;
  int row = blockIdx.x, t = threadIdx.x;
  const float* xr = x + (size_t)row * DIMN;
  float v0 = xr[t], v1 = xr[t + 256], v2 = xr[t + 512];
  sred[t] = v0 + v1 + v2;
  __syncthreads();
  for (int off = 128; off > 0; off >>= 1) {
    if (t < off) sred[t] += sred[t + off];
    __syncthreads();
  }
  if (t == 0) s_mu = sred[0] * (1.0f / DIMN);
  __syncthreads();
  float mu = s_mu;
  float d0 = v0 - mu, d1 = v1 - mu, d2 = v2 - mu;
  sred[t] = d0 * d0 + d1 * d1 + d2 * d2;
  __syncthreads();
  for (int off = 128; off > 0; off >>= 1) {
    if (t < off) sred[t] += sred[t + off];
    __syncthreads();
  }
  if (t == 0) s_rstd = rsqrtf(sred[0] * (1.0f / DIMN) + EPSF);
  __syncthreads();
  float rs = s_rstd;
  _Float16* o = out + (size_t)row * DIMN;
  o[t]       = (_Float16)(d0 * rs * w[t]       + b[t]);
  o[t + 256] = (_Float16)(d1 * rs * w[t + 256] + b[t + 256]);
  o[t + 512] = (_Float16)(d2 * rs * w[t + 512] + b[t + 512]);
}

// ---------------- WMMA GEMM: C[M][768] = A[M][768] @ BT[n][k]^T + bias ----------------
// One wave computes a 16x64 tile (4 accumulators, A fragment reused 4x).
// Epilogue: +bias, optional relu, optional fp32 residual; outputs:
//   outF  : fp32 [m][768]
//   outH  : f16  [m][768]
//   outHT : f16 attention-V layout [(h*B+b)][d][k]  (h=n/64, d=n%64, b=m/2048, k=m%2048)
__global__ void k_gemm(const _Float16* __restrict__ A, const _Float16* __restrict__ BT,
                       const float* __restrict__ bias, const float* __restrict__ resid,
                       float* __restrict__ outF, _Float16* __restrict__ outH,
                       _Float16* __restrict__ outHT, int relu) {
  int lane = threadIdx.x & 31;
  int wid  = blockIdx.x * (blockDim.x >> 5) + (threadIdx.x >> 5);
  int mt = wid / (DIMN / 64);        // 16-row tile index
  int nt = wid % (DIMN / 64);        // 64-col tile index
  int arow = mt * 16 + (lane & 15);
  int nbase = nt * 64;

  v8f acc[4];
#pragma unroll
  for (int j = 0; j < 4; ++j)
#pragma unroll
    for (int e = 0; e < 8; ++e) acc[j][e] = 0.0f;

  for (int kc = 0; kc < DIMN; kc += 32) {
    v16h a = ld_frag(A, DIMN, arow, kc, lane);
#pragma unroll
    for (int j = 0; j < 4; ++j) {
      v16h bt = ld_frag(BT, DIMN, nbase + j * 16 + (lane & 15), kc, lane);
      acc[j] = wmma_f16(a, bt, acc[j]);
    }
  }

#pragma unroll
  for (int j = 0; j < 4; ++j) {
    int n  = nbase + j * 16 + (lane & 15);
    float bv = bias[n];
#pragma unroll
    for (int r = 0; r < 8; ++r) {
      int m = mt * 16 + r + ((lane >> 4) << 3);
      float v = acc[j][r] + bv;
      if (relu) v = fmaxf(v, 0.0f);
      if (resid) v += resid[(size_t)m * DIMN + n];
      if (outF) outF[(size_t)m * DIMN + n] = v;
      if (outH) outH[(size_t)m * DIMN + n] = (_Float16)v;
      if (outHT) {
        int hb = (n >> 6) * BB + (m >> 11);       // h*B + b
        outHT[((size_t)hb * HDIM + (n & 63)) * SS + (m & 2047)] = (_Float16)v;
      }
    }
  }
}

// ---------------- attention: one block per (head, batch, 16-query tile) ----------------
// LDS plan (dynamic): [0,128K) fp32 scores 16x2048 (later reused for PV partials),
//                     [128K,192K) f16 probs 16x2048, then reduction scratch.
#define ATTN_LDS_BYTES (16*SS*4 + 16*SS*2 + (256 + 16 + 16) * 4)

__global__ void k_attn(const _Float16* __restrict__ Q, const _Float16* __restrict__ K,
                       const _Float16* __restrict__ VT, const float* __restrict__ bias,
                       const int* __restrict__ mask, _Float16* __restrict__ attn) {
  extern __shared__ char smem[];
  float*    sc     = (float*)smem;                         // 16 x 2048 f32
  _Float16* pr     = (_Float16*)(smem + 16 * SS * 4);      // 16 x 2048 f16
  float*    red    = (float*)(smem + 16 * SS * 4 + 16 * SS * 2);  // 256
  float*    rowmax = red + 256;                            // 16
  float*    rowsum = rowmax + 16;                          // 16
  float*    pv     = sc;                                   // alias: 8 x 16 x 64 partials

  int lane = threadIdx.x & 31;
  int w    = threadIdx.x >> 5;          // wave 0..7
  int qt   = blockIdx.x & 127;          // query tile
  int hb   = blockIdx.x >> 7;           // h*B + b
  int b    = hb % BB;
  int h    = hb / BB;

  // ---- phase 1: scores = Q Kt + bias, mask, scale; wave w owns keys [w*256, w*256+256) ----
  int qrow = b * SS + qt * 16 + (lane & 15);
  v16h a0 = ld_frag(Q, DIMN, qrow, h * HDIM + 0,  lane);
  v16h a1 = ld_frag(Q, DIMN, qrow, h * HDIM + 32, lane);

  for (int nt = 0; nt < 16; ++nt) {
    int kbase = w * 256 + nt * 16;
    int krow  = b * SS + kbase + (lane & 15);
    v16h b0 = ld_frag(K, DIMN, krow, h * HDIM + 0,  lane);
    v16h b1 = ld_frag(K, DIMN, krow, h * HDIM + 32, lane);
    v8f c;
#pragma unroll
    for (int e = 0; e < 8; ++e) c[e] = 0.0f;
    c = wmma_f16(a0, b0, c);
    c = wmma_f16(a1, b1, c);

    int kcol = kbase + (lane & 15);
#pragma unroll
    for (int r = 0; r < 8; ++r) {
      int m = r + ((lane >> 4) << 3);
      int q = qt * 16 + m;
      // bias/mask are streamed exactly once (450 MB total) -> non-temporal, keep L2
      // for the reused f16 Q/K/V/weight tiles.
      float bsv = __builtin_nontemporal_load(&bias[((size_t)hb * SS + q) * SS + kcol]);
      int   msv = __builtin_nontemporal_load(&mask[((size_t)b  * SS + q) * SS + kcol]);
      float s = c[r] + bsv;
      if (msv == 0) s = -1e20f;
      sc[m * SS + kcol] = s * 0.125f;   // 1/sqrt(HDIM)
    }
  }
  __syncthreads();

  // ---- phase 2: row softmax stats (16 threads per row), unnormalized f16 probs ----
  int row = threadIdx.x >> 4, sub = threadIdx.x & 15;
  float mx = -3.0e38f;
  for (int j = sub; j < SS; j += 16) mx = fmaxf(mx, sc[row * SS + j]);
  red[row * 16 + sub] = mx;
  __syncthreads();
  if (sub == 0) {
    float m2 = -3.0e38f;
#pragma unroll
    for (int i = 0; i < 16; ++i) m2 = fmaxf(m2, red[row * 16 + i]);
    rowmax[row] = m2;
  }
  __syncthreads();
  float rm = rowmax[row];
  float sm = 0.0f;
  for (int j = sub; j < SS; j += 16) {
    float p = __expf(sc[row * SS + j] - rm);
    pr[row * SS + j] = (_Float16)p;
    sm += p;
  }
  red[row * 16 + sub] = sm;
  __syncthreads();
  if (sub == 0) {
    float t = 0.0f;
#pragma unroll
    for (int i = 0; i < 16; ++i) t += red[row * 16 + i];
    rowsum[row] = t;
  }
  __syncthreads();   // after this, sc[] is dead -> safe to alias as pv[]

  // ---- phase 3: PV; wave w accumulates its key slice, partials reduced in LDS ----
  // V is pre-transposed: VT[(h*B+b)][d][k], so B-fragments are contiguous b128 loads.
  const _Float16* vts = VT + (size_t)hb * HDIM * SS;
  v8f o[4];
#pragma unroll
  for (int j = 0; j < 4; ++j)
#pragma unroll
    for (int e = 0; e < 8; ++e) o[j][e] = 0.0f;

  for (int kb = w * 256; kb < w * 256 + 256; kb += 32) {
    v16h pa = ld_frag(pr, SS, (lane & 15), kb, lane);     // probs as A fragment (LDS)
#pragma unroll
    for (int j = 0; j < 4; ++j) {
      v16h vb = ld_frag(vts, SS, j * 16 + (lane & 15), kb, lane);
      o[j] = wmma_f16(pa, vb, o[j]);
    }
  }
#pragma unroll
  for (int j = 0; j < 4; ++j)
#pragma unroll
    for (int r = 0; r < 8; ++r) {
      int m = r + ((lane >> 4) << 3);
      int d = j * 16 + (lane & 15);
      pv[(w * 16 + m) * HDIM + d] = o[j][r];
    }
  __syncthreads();

  for (int idx = threadIdx.x; idx < 16 * HDIM; idx += 256) {
    int m = idx >> 6, d = idx & 63;
    float s = 0.0f;
#pragma unroll
    for (int ww = 0; ww < 8; ++ww) s += pv[(ww * 16 + m) * HDIM + d];
    float val = s / rowsum[m];
    attn[((size_t)(b * SS) + qt * 16 + m) * DIMN + h * HDIM + d] = (_Float16)val;
  }
}

// ---------------- host-side orchestration ----------------
extern "C" void kernel_launch(void* const* d_in, const int* in_sizes, int n_in,
                              void* d_out, int out_size, void* d_ws, size_t ws_size,
                              hipStream_t stream) {
  (void)in_sizes; (void)n_in; (void)out_size; (void)ws_size;
  const float* x     = (const float*)d_in[0];
  const float* bias  = (const float*)d_in[1];
  const int*   mask  = (const int*)  d_in[2];
  const float* ln1w  = (const float*)d_in[3];
  const float* ln1b  = (const float*)d_in[4];
  const float* wq    = (const float*)d_in[5];
  const float* bq    = (const float*)d_in[6];
  const float* wk    = (const float*)d_in[7];
  const float* bk    = (const float*)d_in[8];
  const float* wv    = (const float*)d_in[9];
  const float* bv    = (const float*)d_in[10];
  const float* wo    = (const float*)d_in[11];
  const float* bo    = (const float*)d_in[12];
  const float* ln2w  = (const float*)d_in[13];
  const float* ln2b  = (const float*)d_in[14];
  const float* w1    = (const float*)d_in[15];
  const float* b1    = (const float*)d_in[16];
  const float* w2    = (const float*)d_in[17];
  const float* b2    = (const float*)d_in[18];
  float* out = (float*)d_out;

  // workspace partition (all chunks 256B aligned by construction)
  char* base = (char*)d_ws;
  size_t off = 0;
  const size_t ACT_H = (size_t)NTOK * DIMN * 2;     // f16 activation [4096][768]
  const size_t W_H   = (size_t)DIMN * DIMN * 2;     // f16 weight
  _Float16* xn1 = (_Float16*)(base + off); off += ACT_H;
  _Float16* wT[6];
  for (int i = 0; i < 6; ++i) { wT[i] = (_Float16*)(base + off); off += W_H; }
  _Float16* Qb   = (_Float16*)(base + off); off += ACT_H;
  _Float16* Kb   = (_Float16*)(base + off); off += ACT_H;
  _Float16* VT   = (_Float16*)(base + off); off += ACT_H;    // [(h*B+b)][64][2048]
  _Float16* Ab   = (_Float16*)(base + off); off += ACT_H;    // attention output
  float*    x2   = (float*)   (base + off); off += (size_t)NTOK * DIMN * 4;
  _Float16* xn2  = (_Float16*)(base + off); off += ACT_H;
  _Float16* hb   = (_Float16*)(base + off); off += ACT_H;

  const float* wsrc[6] = {wq, wk, wv, wo, w1, w2};
  dim3 blk(256);
  dim3 gT((DIMN / 32) * (DIMN / 32));
  for (int i = 0; i < 6; ++i)
    k_transpose_cast<<<gT, blk, 0, stream>>>(wsrc[i], wT[i]);

  k_layernorm<<<dim3(NTOK), blk, 0, stream>>>(x, ln1w, ln1b, xn1);

  dim3 gG((NTOK / 16) * (DIMN / 64) / 8);   // 8 waves per block
  k_gemm<<<gG, blk, 0, stream>>>(xn1, wT[0], bq, nullptr, nullptr, Qb, nullptr, 0);
  k_gemm<<<gG, blk, 0, stream>>>(xn1, wT[1], bk, nullptr, nullptr, Kb, nullptr, 0);
  k_gemm<<<gG, blk, 0, stream>>>(xn1, wT[2], bv, nullptr, nullptr, nullptr, VT, 0);

  dim3 gA(HEADS * BB * (SS / 16));          // 3072 blocks
  k_attn<<<gA, blk, ATTN_LDS_BYTES, stream>>>(Qb, Kb, VT, bias, mask, Ab);

  k_gemm<<<gG, blk, 0, stream>>>(Ab, wT[3], bo, x, x2, nullptr, nullptr, 0);

  k_layernorm<<<dim3(NTOK), blk, 0, stream>>>(x2, ln2w, ln2b, xn2);

  k_gemm<<<gG, blk, 0, stream>>>(xn2, wT[4], b1, nullptr, nullptr, hb, nullptr, 1);
  k_gemm<<<gG, blk, 0, stream>>>(hb,  wT[5], b2, x2, out, nullptr, nullptr, 0);
}
